// Model_56727928045875
// MI455X (gfx1250) — compile-verified, matrix operand order checked by reading, not run on hardware
//
#include <hip/hip_runtime.h>
#include <math.h>

// ---------------- problem sizes ----------------
constexpr int Bb = 64, Tt = 600, Dd = 512, Ee = 640, Hh = 1024, Aa = 1024,
              Pp = 1024, Vv = 10025, Nc = 64;
constexpr int K1 = Ee + Dd;        // 1152  (x = [emb, ctx])
constexpr int K2 = Hh + Ee + Dd;   // 2176  (readout input)
constexpr int Gg = 4 * Hh;         // 4096  (gates)
constexpr int PH = Pp / 2;         // 512   (maxout output)
constexpr float ZONE = 0.05f;

typedef __bf16 bf16_t;
typedef __attribute__((ext_vector_type(16))) __bf16 v16bf;
typedef __attribute__((ext_vector_type(8)))  __bf16 v8bf;
typedef __attribute__((ext_vector_type(4)))  __bf16 v4bf;
typedef __attribute__((ext_vector_type(8)))  float  v8f;

static __device__ __forceinline__ bf16_t f2bf(float f) {
  unsigned u = __builtin_bit_cast(unsigned, f);
  unsigned r = u + 0x7FFFu + ((u >> 16) & 1u);          // round-to-nearest-even
  unsigned short s = (unsigned short)(r >> 16);
  return __builtin_bit_cast(bf16_t, s);
}
static __device__ __forceinline__ float bf2f(bf16_t b) {
  unsigned short s = __builtin_bit_cast(unsigned short, b);
  unsigned u = ((unsigned)s) << 16;
  return __builtin_bit_cast(float, u);
}
static __device__ __forceinline__ float sigm(float x) { return 1.f / (1.f + expf(-x)); }

// ---------------- utility kernels ----------------
// counts are all multiples of 4 -> vectorized float4 -> packed 4xbf16
__global__ void __launch_bounds__(256) cvt_kernel(const float* __restrict__ in,
                                                  bf16_t* __restrict__ out, long long n4) {
  for (long long i = (long long)blockIdx.x * 256 + threadIdx.x; i < n4;
       i += (long long)gridDim.x * 256) {
    float4 v = ((const float4*)in)[i];
    v4bf o;
    o[0] = f2bf(v.x); o[1] = f2bf(v.y); o[2] = f2bf(v.z); o[3] = f2bf(v.w);
    ((v4bf*)out)[i] = o;
  }
}

__global__ void __launch_bounds__(256) bias_sum_kernel(const float* a, const float* b,
                                                       float* out, int n) {
  int i = blockIdx.x * 256 + threadIdx.x;
  if (i < n) out[i] = a[i] + b[i];
}

// shifted target embeddings: emb[b,0,:]=0 ; emb[b,n,:]=embed[labels[b,n-1]]
__global__ void __launch_bounds__(256) embed_kernel(const float* __restrict__ embed,
                                                    const int* __restrict__ labels,
                                                    float* __restrict__ emb_f,
                                                    bf16_t* __restrict__ emb_bf) {
  long long total = (long long)Bb * Nc * Ee;
  for (long long i = (long long)blockIdx.x * 256 + threadIdx.x; i < total;
       i += (long long)gridDim.x * 256) {
    int e  = (int)(i % Ee);
    int bn = (int)(i / Ee);
    int n  = bn % Nc, b = bn / Nc;
    float v = 0.f;
    if (n > 0) {
      int lbl = labels[b * Nc + n - 1];
      v = embed[(size_t)lbl * Ee + e];
    }
    emb_f[i]  = v;
    emb_bf[i] = f2bf(v);
  }
}

// ---------------- WMMA GEMM:  C[M,N] = A[M,K] @ W[N,K]^T (+bias) ----------------
// One wave -> 32x64 strip: 2 A fragments x 4 B fragments -> 8 wmma per K-step,
// B fragments reused across both M tiles (24 B loaded / wmma).
// OOB W rows are CLAMPED to N-1: row r of a W tile only feeds output column
// n0+j*16+r, which the store guard drops when OOB -> no masking in the K loop.
static __device__ __forceinline__ v16bf load_frag16(const bf16_t* p) {
  v8bf lo = *(const v8bf*)(p);        // K = k .. k+7     (lanes>=16: +8..)
  v8bf hi = *(const v8bf*)(p + 16);   // K = k+16 .. k+23 (lanes>=16: +24..)
  v16bf r;
#pragma unroll
  for (int i = 0; i < 8; ++i) { r[i] = lo[i]; r[i + 8] = hi[i]; }
  return r;
}

template <bool STORE_BF16>
__global__ void __launch_bounds__(256)
gemm_wmma(const bf16_t* __restrict__ Amat, const bf16_t* __restrict__ Wmat,
          const float* __restrict__ bias, void* __restrict__ Cout,
          int M, int N, int K) {
  const int lane = threadIdx.x & 31;
  const int wave = threadIdx.x >> 5;
  const int ntile = (N + 63) >> 6;
  const int task = blockIdx.x * 8 + wave;
  const int tm = task / ntile;
  if (tm * 32 >= M) return;
  const int n0 = (task - tm * ntile) * 64;
  const int m0 = tm * 32;
  const int row  = lane & 15;
  const int koff = (lane >> 4) << 3;   // 0 or 8

  v8f acc0[4] = {{}, {}, {}, {}};
  v8f acc1[4] = {{}, {}, {}, {}};
  const bf16_t* arow0 = Amat + (size_t)(m0 + row) * K + koff;
  const bf16_t* arow1 = arow0 + (size_t)16 * K;
  const bf16_t* wp[4];
#pragma unroll
  for (int j = 0; j < 4; ++j) {
    int nr = n0 + j * 16 + row;
    if (nr > N - 1) nr = N - 1;        // clamp (see comment above)
    wp[j] = Wmat + (size_t)nr * K + koff;
  }

  for (int k = 0; k < K; k += 32) {
    v16bf a0 = load_frag16(arow0 + k);
    v16bf a1 = load_frag16(arow1 + k);
    __builtin_prefetch(arow0 + k + 1024, 0, 1);   // global_prefetch_b8
#pragma unroll
    for (int j = 0; j < 4; ++j) {
      v16bf bfr = load_frag16(wp[j] + k);
      acc0[j] = __builtin_amdgcn_wmma_f32_16x16x32_bf16(
          false, a0, false, bfr, (short)0, acc0[j], false, false);
      acc1[j] = __builtin_amdgcn_wmma_f32_16x16x32_bf16(
          false, a1, false, bfr, (short)0, acc1[j], false, false);
    }
  }

  const int rowoff = (lane >> 4) << 3;  // lanes 16-31 hold M = r+8
#pragma unroll
  for (int j = 0; j < 4; ++j) {
    int col = n0 + j * 16 + (lane & 15);
    if (col >= N) continue;
    float badd = bias ? bias[col] : 0.f;
#pragma unroll
    for (int rr = 0; rr < 8; ++rr) {
      size_t off0 = (size_t)(m0 + rr + rowoff) * (size_t)N + col;
      size_t off1 = (size_t)(m0 + 16 + rr + rowoff) * (size_t)N + col;
      float v0 = acc0[j][rr] + badd;
      float v1 = acc1[j][rr] + badd;
      if (STORE_BF16) {
        ((bf16_t*)Cout)[off0] = f2bf(v0);
        ((bf16_t*)Cout)[off1] = f2bf(v1);
      } else {
        ((float*)Cout)[off0] = v0;
        ((float*)Cout)[off1] = v1;
      }
    }
  }
}

// ---------------- inv_fertility: sigmoid(enc[b,t,:] . W_fert), wave per row ----------------
__global__ void __launch_bounds__(256) invfert_kernel(const float* __restrict__ enc,
                                                      const float* __restrict__ wf,
                                                      float* __restrict__ out) {
  int lane = threadIdx.x & 31;
  int rowi = blockIdx.x * 8 + (threadIdx.x >> 5);
  if (rowi >= Bb * Tt) return;
  const float* r = enc + (size_t)rowi * Dd;
  float s = 0.f;
  for (int d = lane * 4; d < Dd; d += 128) {
    float4 v = *(const float4*)(r + d);
    float4 c = *(const float4*)(wf + d);
    s += v.x * c.x + v.y * c.y + v.z * c.z + v.w * c.w;
  }
  for (int off = 16; off; off >>= 1) s += __shfl_xor(s, off);
  if (lane == 0) out[rowi] = sigm(s);
}

// ---------------- attention energies ----------------
// e[b,t] = sum_a v[a] * tanh(enc_ctx[b,t,a] + q[b,a] + accum[b,t]*W_fb[a]); mask -> -inf
// enc_ctx is the per-step hot stream (75MB, L2-resident): b128 vector loads.
__global__ void __launch_bounds__(256) attn_kernel(const bf16_t* __restrict__ encctx,
                                                   const float* __restrict__ q,
                                                   const float* __restrict__ accum,
                                                   const float* __restrict__ vatt,
                                                   const float* __restrict__ wfb,
                                                   const int* __restrict__ slen,
                                                   float* __restrict__ e) {
  __shared__ float q_s[Aa], v_s[Aa], f_s[Aa];
  const int blocksPerB = Tt / 8;
  int b  = blockIdx.x / blocksPerB;
  int t0 = (blockIdx.x % blocksPerB) * 8;
  for (int i = threadIdx.x; i < Aa; i += 256) {
    q_s[i] = q[b * Aa + i];
    v_s[i] = vatt[i];
    f_s[i] = wfb[i];
  }
  __syncthreads();
  int lane = threadIdx.x & 31;
  int t = t0 + (threadIdx.x >> 5);
  float ac = accum[b * Tt + t];
  const bf16_t* rowp = encctx + ((size_t)b * Tt + t) * Aa;
  float s = 0.f;
  for (int a0 = lane * 8; a0 < Aa; a0 += 256) {   // 4 x b128 per lane
    v8bf v = *(const v8bf*)(rowp + a0);
#pragma unroll
    for (int i = 0; i < 8; ++i) {
      int a = a0 + i;
      float x = bf2f(v[i]) + q_s[a] + ac * f_s[a];
      s += v_s[a] * tanhf(x);
    }
  }
  for (int off = 16; off; off >>= 1) s += __shfl_xor(s, off);
  if (lane == 0) e[b * Tt + t] = (t < slen[b]) ? s : -__builtin_inff();
}

// ---------------- softmax over T + accum update (one block per batch) ----------------
__global__ void __launch_bounds__(256) softmax_kernel(const float* __restrict__ e,
                                                      const float* __restrict__ invf,
                                                      float* __restrict__ w,
                                                      float* __restrict__ accum) {
  int b = blockIdx.x;
  __shared__ float red[8];
  __shared__ float s_max, s_sum;
  int lane = threadIdx.x & 31, wv = threadIdx.x >> 5;

  float m = -__builtin_inff();
  for (int t = threadIdx.x; t < Tt; t += 256) m = fmaxf(m, e[b * Tt + t]);
  for (int off = 16; off; off >>= 1) m = fmaxf(m, __shfl_xor(m, off));
  if (lane == 0) red[wv] = m;
  __syncthreads();
  if (threadIdx.x == 0) {
    float mm = red[0];
    for (int i = 1; i < 8; ++i) mm = fmaxf(mm, red[i]);
    s_max = mm;
  }
  __syncthreads();
  m = s_max;

  float s = 0.f;
  for (int t = threadIdx.x; t < Tt; t += 256) s += expf(e[b * Tt + t] - m);
  for (int off = 16; off; off >>= 1) s += __shfl_xor(s, off);
  __syncthreads();
  if (lane == 0) red[wv] = s;
  __syncthreads();
  if (threadIdx.x == 0) {
    float ss = 0.f;
    for (int i = 0; i < 8; ++i) ss += red[i];
    s_sum = ss;
  }
  __syncthreads();
  float inv = 1.f / s_sum;
  for (int t = threadIdx.x; t < Tt; t += 256) {
    float wt = expf(e[b * Tt + t] - m) * inv;   // masked slots -> exp(-inf)=0
    w[b * Tt + t] = wt;
    accum[b * Tt + t] += wt * invf[b * Tt + t] * 0.5f;
  }
}

// ---------------- context: ctx[b,:] = sum_t w[b,t] * enc[b,t,:] ----------------
// 4 t-groups x 64 lanes x 8-wide b128 loads; LDS tree reduction across groups.
__global__ void __launch_bounds__(256) ctx_kernel(const bf16_t* __restrict__ enc_bf,
                                                  const float* __restrict__ w,
                                                  float* __restrict__ ctx,
                                                  float* __restrict__ ctx_seq, int n) {
  int b = blockIdx.x;
  __shared__ float ws_[Tt];
  __shared__ float part[4][Dd];
  for (int t = threadIdx.x; t < Tt; t += 256) ws_[t] = w[b * Tt + t];
  __syncthreads();
  int grp = threadIdx.x >> 6, lid = threadIdx.x & 63;
  int d0 = lid * 8;
  float acc[8];
#pragma unroll
  for (int i = 0; i < 8; ++i) acc[i] = 0.f;
  const bf16_t* base = enc_bf + (size_t)b * Tt * Dd + d0;
  for (int t = grp; t < Tt; t += 4) {
    float wt = ws_[t];
    v8bf v = *(const v8bf*)(base + (size_t)t * Dd);
#pragma unroll
    for (int i = 0; i < 8; ++i) acc[i] += wt * bf2f(v[i]);
  }
#pragma unroll
  for (int i = 0; i < 8; ++i) part[grp][d0 + i] = acc[i];
  __syncthreads();
  if (grp == 0) {
    size_t o = ((size_t)b * Nc + n) * Dd;
#pragma unroll
    for (int i = 0; i < 8; ++i) {
      float s = part[0][d0 + i] + part[1][d0 + i] + part[2][d0 + i] + part[3][d0 + i];
      ctx[b * Dd + d0 + i] = s;
      ctx_seq[o + d0 + i]  = s;
    }
  }
}

// ---------------- build x = [emb_t, ctx] in bf16 ----------------
__global__ void __launch_bounds__(256) buildx_kernel(const bf16_t* __restrict__ emb_bf,
                                                     const float* __restrict__ ctx,
                                                     bf16_t* __restrict__ x, int n) {
  int i = blockIdx.x * 256 + threadIdx.x;
  if (i >= Bb * K1) return;
  int k = i % K1, b = i / K1;
  if (k < Ee) x[i] = emb_bf[((size_t)b * Nc + n) * Ee + k];
  else        x[i] = f2bf(ctx[b * Dd + (k - Ee)]);
}

// ---------------- LSTM cell activations (zero state) + zoneout on h ----------------
__global__ void __launch_bounds__(256) lstm_kernel(const float* __restrict__ gates,
                                                   float* __restrict__ h,
                                                   bf16_t* __restrict__ h_bf,
                                                   float* __restrict__ s_seq, int n) {
  int i = blockIdx.x * 256 + threadIdx.x;
  if (i >= Bb * Hh) return;
  int hh = i % Hh, b = i / Hh;
  const float* g = gates + (size_t)b * Gg;
  float gi = g[hh], gg = g[2 * Hh + hh], go = g[3 * Hh + hh];
  float cn = sigm(gi) * tanhf(gg);       // f-gate * 0 dropped (zero state)
  float hn = sigm(go) * tanhf(cn);
  float hv = ZONE * h[i] + (1.f - ZONE) * hn;   // zoneout interpolation
  h[i] = hv;
  h_bf[i] = f2bf(hv);
  s_seq[((size_t)b * Nc + n) * Hh + hh] = hv;   // c-state is dead downstream: elided
}

// ---------------- readout input concat [s, emb, ctx] -> bf16 ----------------
__global__ void __launch_bounds__(256) rin_kernel(const float* __restrict__ s_seq,
                                                  const float* __restrict__ emb_f,
                                                  const float* __restrict__ ctx_seq,
                                                  bf16_t* __restrict__ rin) {
  long long total = (long long)Bb * Nc * K2;
  for (long long i = (long long)blockIdx.x * 256 + threadIdx.x; i < total;
       i += (long long)gridDim.x * 256) {
    int k = (int)(i % K2);
    long long rowl = i / K2;
    float v;
    if (k < Hh)            v = s_seq[rowl * Hh + k];
    else if (k < Hh + Ee)  v = emb_f[rowl * Ee + (k - Hh)];
    else                   v = ctx_seq[rowl * Dd + (k - Hh - Ee)];
    rin[i] = f2bf(v);
  }
}

// ---------------- maxout(2) -> bf16 ----------------
__global__ void __launch_bounds__(256) maxout_kernel(const float* __restrict__ r,
                                                     bf16_t* __restrict__ rh) {
  long long total = (long long)Bb * Nc * PH;
  for (long long i = (long long)blockIdx.x * 256 + threadIdx.x; i < total;
       i += (long long)gridDim.x * 256) {
    int p = (int)(i % PH);
    long long rowl = i / PH;
    const float* rr = r + rowl * Pp;
    rh[i] = f2bf(fmaxf(rr[2 * p], rr[2 * p + 1]));
  }
}

// ---------------- host ----------------
extern "C" void kernel_launch(void* const* d_in, const int* in_sizes, int n_in,
                              void* d_out, int out_size, void* d_ws, size_t ws_size,
                              hipStream_t stream) {
  const float* enc    = (const float*)d_in[0];
  const int*   labels = (const int*)d_in[1];
  const int*   slen   = (const int*)d_in[2];
  const float* embed  = (const float*)d_in[3];
  const float* W_ih   = (const float*)d_in[4];
  const float* b_ih   = (const float*)d_in[5];
  const float* b_hh   = (const float*)d_in[6];
  const float* W_s    = (const float*)d_in[7];
  const float* W_enc  = (const float*)d_in[8];
  const float* b_enc  = (const float*)d_in[9];
  const float* v_att  = (const float*)d_in[10];
  const float* W_fert = (const float*)d_in[11];
  const float* W_fb   = (const float*)d_in[12];
  const float* W_r    = (const float*)d_in[13];
  const float* b_r    = (const float*)d_in[14];
  const float* W_o    = (const float*)d_in[15];
  const float* b_o    = (const float*)d_in[16];
  float* logits = (float*)d_out;

  char* base = (char*)d_ws;
  size_t off = 0;
  auto alloc = [&](size_t bytes) {
    size_t o = off;
    off += (bytes + 255) & ~(size_t)255;
    return o;
  };
  bf16_t* enc_bf    = (bf16_t*)(base + alloc((size_t)Bb * Tt * Dd * 2));
  bf16_t* Wih_bf    = (bf16_t*)(base + alloc((size_t)Gg * K1 * 2));
  bf16_t* Ws_bf     = (bf16_t*)(base + alloc((size_t)Aa * Hh * 2));
  bf16_t* Wenc_bf   = (bf16_t*)(base + alloc((size_t)Aa * Dd * 2));
  bf16_t* Wr_bf     = (bf16_t*)(base + alloc((size_t)Pp * K2 * 2));
  bf16_t* Wo_bf     = (bf16_t*)(base + alloc((size_t)Vv * PH * 2));
  float*  bias_g    = (float*)(base + alloc((size_t)Gg * 4));
  float*  emb_f     = (float*)(base + alloc((size_t)Bb * Nc * Ee * 4));
  bf16_t* emb_bf    = (bf16_t*)(base + alloc((size_t)Bb * Nc * Ee * 2));
  bf16_t* encctx_bf = (bf16_t*)(base + alloc((size_t)Bb * Tt * Aa * 2));
  float*  invfert   = (float*)(base + alloc((size_t)Bb * Tt * 4));
  float*  accum     = (float*)(base + alloc((size_t)Bb * Tt * 4));
  float*  hbuf      = (float*)(base + alloc((size_t)Bb * Hh * 4));
  bf16_t* hbf       = (bf16_t*)(base + alloc((size_t)Bb * Hh * 2));
  float*  ctx       = (float*)(base + alloc((size_t)Bb * Dd * 4));
  bf16_t* xbf       = (bf16_t*)(base + alloc((size_t)Bb * K1 * 2));
  float*  gates     = (float*)(base + alloc((size_t)Bb * Gg * 4));
  float*  qbuf      = (float*)(base + alloc((size_t)Bb * Aa * 4));
  float*  ebuf      = (float*)(base + alloc((size_t)Bb * Tt * 4));
  float*  wbuf      = (float*)(base + alloc((size_t)Bb * Tt * 4));
  float*  s_seq     = (float*)(base + alloc((size_t)Bb * Nc * Hh * 4));
  float*  ctx_seq   = (float*)(base + alloc((size_t)Bb * Nc * Dd * 4));
  bf16_t* rin       = (bf16_t*)(base + alloc((size_t)Bb * Nc * K2 * 2));
  float*  rbuf      = (float*)(base + alloc((size_t)Bb * Nc * Pp * 4));
  bf16_t* rhbf      = (bf16_t*)(base + alloc((size_t)Bb * Nc * PH * 2));
  (void)ws_size; (void)in_sizes; (void)n_in; (void)out_size;

  auto cvt = [&](const float* in, bf16_t* out, long long n) {
    long long n4 = n >> 2;                      // all counts are multiples of 4
    int blocks = (int)((n4 + 255) / 256);
    cvt_kernel<<<blocks, 256, 0, stream>>>(in, out, n4);
  };
  auto gemm = [&](const bf16_t* Am, const bf16_t* Wm, const float* bias, void* C,
                  int M, int Nd, int K, bool tobf) {
    int tasks = (M / 32) * ((Nd + 63) / 64);
    int blocks = (tasks + 7) / 8;
    if (tobf)
      gemm_wmma<true><<<blocks, 256, 0, stream>>>(Am, Wm, bias, C, M, Nd, K);
    else
      gemm_wmma<false><<<blocks, 256, 0, stream>>>(Am, Wm, bias, C, M, Nd, K);
  };

  // ---- one-time prep ----
  cvt(enc,   enc_bf,  (long long)Bb * Tt * Dd);
  cvt(W_ih,  Wih_bf,  (long long)Gg * K1);
  cvt(W_s,   Ws_bf,   (long long)Aa * Hh);
  cvt(W_enc, Wenc_bf, (long long)Aa * Dd);
  cvt(W_r,   Wr_bf,   (long long)Pp * K2);
  cvt(W_o,   Wo_bf,   (long long)Vv * PH);
  bias_sum_kernel<<<(Gg + 255) / 256, 256, 0, stream>>>(b_ih, b_hh, bias_g, Gg);
  embed_kernel<<<4096, 256, 0, stream>>>(embed, labels, emb_f, emb_bf);

  // enc_ctx = enc @ W_enc^T + b_enc  (stored bf16, L2-resident for the scan)
  gemm(enc_bf, Wenc_bf, b_enc, encctx_bf, Bb * Tt, Aa, Dd, true);
  invfert_kernel<<<(Bb * Tt + 7) / 8, 256, 0, stream>>>(enc, W_fert, invfert);

  hipMemsetAsync(accum, 0, (size_t)Bb * Tt * 4, stream);
  hipMemsetAsync(hbuf,  0, (size_t)Bb * Hh * 4, stream);
  hipMemsetAsync(ctx,   0, (size_t)Bb * Dd * 4, stream);

  // ---- sequential decoder scan ----
  for (int n = 0; n < Nc; ++n) {
    buildx_kernel<<<(Bb * K1 + 255) / 256, 256, 0, stream>>>(emb_bf, ctx, xbf, n);
    gemm(xbf, Wih_bf, bias_g, gates, Bb, Gg, K1, false);
    lstm_kernel<<<(Bb * Hh + 255) / 256, 256, 0, stream>>>(gates, hbuf, hbf, s_seq, n);
    gemm(hbf, Ws_bf, nullptr, qbuf, Bb, Aa, Hh, false);
    attn_kernel<<<Bb * (Tt / 8), 256, 0, stream>>>(encctx_bf, qbuf, accum, v_att,
                                                   W_fb, slen, ebuf);
    softmax_kernel<<<Bb, 256, 0, stream>>>(ebuf, invfert, wbuf, accum);
    ctx_kernel<<<Bb, 256, 0, stream>>>(enc_bf, wbuf, ctx, ctx_seq, n);
  }

  // ---- readout ----
  rin_kernel<<<8192, 256, 0, stream>>>(s_seq, emb_f, ctx_seq, rin);
  gemm(rin, Wr_bf, b_r, rbuf, Bb * Nc, Pp, K2, false);
  maxout_kernel<<<4096, 256, 0, stream>>>(rbuf, rhbf);
  gemm(rhbf, Wo_bf, b_o, logits, Bb * Nc, Vv, PH, false);
}
